// GraphTemporalEmbedding_88914412962282
// MI455X (gfx1250) — matrix operand order repaired and checked by reading, not compiled, not run on hardware
//
#include <hip/hip_runtime.h>
#include <hip/hip_bf16.h>
#include <stdint.h>

// ---------------------------------------------------------------------------
// CDNA5 / gfx1250: wave32, WMMA 16x16x32 bf16 (f32 accumulate)
// Each wave computes a 64(M) x 32(N) output tile: 8 WMMAs per K=32 chunk,
// A-fragments reused 2x, B-fragments 4x -> ~0.67 WMMA per b128 load.
// ---------------------------------------------------------------------------
typedef __bf16 bf16_t;
typedef __attribute__((ext_vector_type(16))) __bf16 v16bf;
typedef __attribute__((ext_vector_type(8)))  float  v8f;

static constexpr int Bz = 16;    // batch
static constexpr int Nc = 512;   // channels / graph nodes
static constexpr int Sp = 1024;  // sequence length
static constexpr int Lx = 3;     // layers
static constexpr int Kw = 3;     // conv kernel width

// ---- bf16 <-> f32 via bit ops (RNE), independent of toolchain cast support
__device__ __forceinline__ bf16_t f2bf(float f) {
  union { float f; uint32_t u; } in; in.f = f;
  uint32_t u = in.u + 0x7FFFu + ((in.u >> 16) & 1u);
  union { uint16_t s; bf16_t b; } out; out.s = (uint16_t)(u >> 16);
  return out.b;
}
__device__ __forceinline__ float bf2f(bf16_t b) {
  union { bf16_t b; uint16_t s; } in; in.b = b;
  union { uint32_t u; float f; } out; out.u = ((uint32_t)in.s) << 16;
  return out.f;
}

__device__ __forceinline__ v8f wmma_bf16(v16bf a, v16bf b, v8f c) {
  // 8 args: (neg_a, A, neg_b, B, c_mod, C, reuse_a, reuse_b)
  return __builtin_amdgcn_wmma_f32_16x16x32_bf16(false, a, false, b, (short)0, c,
                                                 false, false);
}

union Frag  { v16bf v; uint4 q[2]; };
union Pack8 { uint4 q; bf16_t e[8]; };
union PackF { float4 q[2]; float f[8]; };

// A fragment (16x32, row-major): lane half=0 -> K[0..8)+K[16..24);
// half=1 -> K[8..16)+K[24..32); row m = lane%16.
__device__ __forceinline__ v16bf load_a_row(const bf16_t* __restrict__ row,
                                            int kbase, int half) {
  Frag f;
  f.q[0] = *(const uint4*)(row + kbase + half * 8);
  f.q[1] = *(const uint4*)(row + kbase + 16 + half * 8);
  return f.v;
}
// B fragment (32x16): column n = lane%16; half=0 -> K[0..8)+K[16..24); half=1 -> K[8..16)+K[24..32)
__device__ __forceinline__ v16bf load_b_col(const bf16_t* __restrict__ col,
                                            int kbase, int half) {
  Frag f;
  const bf16_t* p = col + kbase + half * 8;
  f.q[0] = *(const uint4*)(p);
  f.q[1] = *(const uint4*)(p + 16);
  return f.v;
}

// ---------------------------------------------------------------------------
// Prep kernels (elementwise; cheap vs. the GEMM work)
// ---------------------------------------------------------------------------
__global__ void cvt_f32_to_bf16(const float* __restrict__ in,
                                bf16_t* __restrict__ out, int n) {
  int i = blockIdx.x * blockDim.x + threadIdx.x;
  if (i < n) out[i] = f2bf(in[i]);
}

// conv_w [L,N,N,K] (o,i,k) -> [L,K,O,I] bf16, row o contiguous over i
__global__ void prep_conv_w(const float* __restrict__ w,
                            bf16_t* __restrict__ out, int total) {
  int idx = blockIdx.x * blockDim.x + threadIdx.x;
  if (idx >= total) return;
  int i   = idx % Nc; int rem = idx / Nc;
  int o   = rem % Nc; rem /= Nc;
  int k   = rem % Kw; int l = rem / Kw;
  out[idx] = f2bf(w[(((size_t)(l * Nc + o)) * Nc + i) * Kw + k]);
}

// gc_weight [S,S] (t,t') -> gcT [t'][t] bf16 (t contiguous)
__global__ void prep_gcT(const float* __restrict__ gc,
                         bf16_t* __restrict__ out, int total) {
  int idx = blockIdx.x * blockDim.x + threadIdx.x;
  if (idx >= total) return;
  int t  = idx % Sp;
  int tp = idx / Sp;
  out[idx] = f2bf(gc[(size_t)t * Sp + tp]);
}

// hard Gumbel z0/N: forward value = one_hot(argmax(logits+g))[...,0] / N
__global__ void prep_adj(const float* __restrict__ logits,
                         const float* __restrict__ gn,
                         bf16_t* __restrict__ out, int total) {
  int idx = blockIdx.x * blockDim.x + threadIdx.x;
  if (idx >= total) return;
  int e = idx % (Nc * Nc);
  int l = idx / (Nc * Nc);
  size_t gbase = ((size_t)l * Nc * Nc + e) * 2;
  float a0 = logits[2 * e + 0] + gn[gbase + 0];
  float a1 = logits[2 * e + 1] + gn[gbase + 1];
  out[idx] = f2bf((a0 >= a1) ? (1.0f / (float)Nc) : 0.0f);  // 1/512 is bf16-exact
}

// ---------------------------------------------------------------------------
// Dilated circular conv as WMMA GEMM.
// D[m=o, n=t] = sum_k sum_i w[k][o][i] * h[b, (t+(k-1)d) & (S-1), i]
// Wave tile: 64 o x 32 t. out stored [B,T,C] bf16; optional residual
// (read at own tile only -> in-place update of h is safe).
// ---------------------------------------------------------------------------
template <bool RES>
__global__ __launch_bounds__(32)
void conv_wmma(const bf16_t* __restrict__ hin, const bf16_t* __restrict__ wk,
               const float* __restrict__ bias, const bf16_t* __restrict__ hres,
               bf16_t* __restrict__ out, int dil) {
  const int lane = threadIdx.x;
  const int n15  = lane & 15;
  const int half = lane >> 4;
  const int b  = blockIdx.z;
  const int o0 = blockIdx.y * 64;
  const int t0 = blockIdx.x * 32;
  const bf16_t* hb = hin + (size_t)b * Sp * Nc;

  v8f acc[4][2] = {};  // [o-subtile][t-subtile]
  for (int k = 0; k < Kw; ++k) {
    const int shift = (k - 1) * dil;
    const bf16_t* w = wk + (size_t)k * Nc * Nc;
    const bf16_t* rowA[4];
#pragma unroll
    for (int s = 0; s < 4; ++s)
      rowA[s] = w + (size_t)(o0 + 16 * s + n15) * Nc;
    const int tc0 = (t0 + n15 + shift) & (Sp - 1);
    const int tc1 = (t0 + 16 + n15 + shift) & (Sp - 1);
    const bf16_t* col0 = hb + (size_t)tc0 * Nc;
    const bf16_t* col1 = hb + (size_t)tc1 * Nc;
    for (int kb = 0; kb < Nc; kb += 32) {
      v16bf bf0 = load_b_col(col0, kb, half);
      v16bf bf1 = load_b_col(col1, kb, half);
#pragma unroll
      for (int s = 0; s < 4; ++s) {
        v16bf a = load_a_row(rowA[s], kb, half);
        acc[s][0] = wmma_bf16(a, bf0, acc[s][0]);
        acc[s][1] = wmma_bf16(a, bf1, acc[s][1]);
      }
    }
  }
  // epilogue: element (m = r + 8*half, n = n15); store [T,C] packed 8xbf16
#pragma unroll
  for (int it = 0; it < 2; ++it) {
    const int t = t0 + it * 16 + n15;
    bf16_t* orow = out + ((size_t)b * Sp + t) * Nc + o0;
    const bf16_t* rrow = hres + ((size_t)b * Sp + t) * Nc + o0;
#pragma unroll
    for (int io = 0; io < 4; ++io) {
      const int obase = io * 16 + 8 * half;
      Pack8 rv;
      if (RES) rv.q = *(const uint4*)(rrow + obase);
      Pack8 pk;
#pragma unroll
      for (int r = 0; r < 8; ++r) {
        float v = acc[io][it][r] + bias[o0 + obase + r];
        if (RES) v += bf2f(rv.e[r]);
        v = v > 0.f ? v : 0.f;
        pk.e[r] = f2bf(v);
      }
      *(uint4*)(orow + obase) = pk.q;
    }
  }
}

// ---------------------------------------------------------------------------
// GCN stage 1: U[b,i,t] = sum_j A[i,j] * h[b,t,j]
// WMMA: D[m=t, n=i] = sum_j h[t,j] * A[i,j]; both operands j-contiguous.
// Wave tile: 64 t x 32 i. U stored [B,C,T] bf16 (t contiguous packed stores).
// ---------------------------------------------------------------------------
__global__ __launch_bounds__(32)
void gcn_aggregate(const bf16_t* __restrict__ h, const bf16_t* __restrict__ Am,
                   bf16_t* __restrict__ U) {
  const int lane = threadIdx.x;
  const int n15  = lane & 15;
  const int half = lane >> 4;
  const int b  = blockIdx.z;
  const int i0 = blockIdx.y * 32;
  const int t0 = blockIdx.x * 64;
  const bf16_t* hb = h + (size_t)b * Sp * Nc;
  const bf16_t* rowA[4];
#pragma unroll
  for (int s = 0; s < 4; ++s)
    rowA[s] = hb + (size_t)(t0 + 16 * s + n15) * Nc;
  const bf16_t* colB0 = Am + (size_t)(i0 + n15) * Nc;
  const bf16_t* colB1 = Am + (size_t)(i0 + 16 + n15) * Nc;

  v8f acc[4][2] = {};  // [t-subtile(m)][i-subtile(n)]
  for (int kb = 0; kb < Nc; kb += 32) {
    v16bf bf0 = load_b_col(colB0, kb, half);
    v16bf bf1 = load_b_col(colB1, kb, half);
#pragma unroll
    for (int s = 0; s < 4; ++s) {
      v16bf a = load_a_row(rowA[s], kb, half);
      acc[s][0] = wmma_bf16(a, bf0, acc[s][0]);
      acc[s][1] = wmma_bf16(a, bf1, acc[s][1]);
    }
  }
#pragma unroll
  for (int in_ = 0; in_ < 2; ++in_) {
    const int i = i0 + in_ * 16 + n15;
    bf16_t* up = U + ((size_t)b * Nc + i) * Sp + t0;
#pragma unroll
    for (int im = 0; im < 4; ++im) {
      const int tbase = im * 16 + 8 * half;
      Pack8 pk;
#pragma unroll
      for (int r = 0; r < 8; ++r) pk.e[r] = f2bf(acc[im][in_][r]);
      *(uint4*)(up + tbase) = pk.q;
    }
  }
}

// ---------------------------------------------------------------------------
// GCN stage 2: h[b,i,t'] = sum_t U[b,i,t] * gc[t,t'] + gc_bias[t']
// WMMA: D[m=i, n=t'] = sum_t U_ct[i,t] * gcT[t',t]; both operands t-contiguous.
// Wave tile: 64 i x 32 t'. Output back to [B,T,C] (== final transpose).
// Last layer writes f32 directly to d_out.
// ---------------------------------------------------------------------------
template <bool LAST>
__global__ __launch_bounds__(32)
void gcn_mix(const bf16_t* __restrict__ U, const bf16_t* __restrict__ gcT,
             const float* __restrict__ gcb, bf16_t* __restrict__ hout,
             float* __restrict__ fout) {
  const int lane = threadIdx.x;
  const int n15  = lane & 15;
  const int half = lane >> 4;
  const int b   = blockIdx.z;
  const int i0  = blockIdx.y * 64;
  const int tp0 = blockIdx.x * 32;
  const bf16_t* Ub = U + (size_t)b * Nc * Sp;
  const bf16_t* rowA[4];
#pragma unroll
  for (int s = 0; s < 4; ++s)
    rowA[s] = Ub + (size_t)(i0 + 16 * s + n15) * Sp;
  const bf16_t* colB0 = gcT + (size_t)(tp0 + n15) * Sp;
  const bf16_t* colB1 = gcT + (size_t)(tp0 + 16 + n15) * Sp;

  v8f acc[4][2] = {};  // [i-subtile(m)][t'-subtile(n)]
  for (int kb = 0; kb < Sp; kb += 32) {
    v16bf bf0 = load_b_col(colB0, kb, half);
    v16bf bf1 = load_b_col(colB1, kb, half);
#pragma unroll
    for (int s = 0; s < 4; ++s) {
      v16bf a = load_a_row(rowA[s], kb, half);
      acc[s][0] = wmma_bf16(a, bf0, acc[s][0]);
      acc[s][1] = wmma_bf16(a, bf1, acc[s][1]);
    }
  }
#pragma unroll
  for (int it = 0; it < 2; ++it) {
    const int tp = tp0 + it * 16 + n15;
    const float bb = gcb[tp];
    if (LAST) {
      float* orow = fout + ((size_t)b * Sp + tp) * Nc + i0;
#pragma unroll
      for (int io = 0; io < 4; ++io) {
        const int ibase = io * 16 + 8 * half;
        PackF pf;
#pragma unroll
        for (int r = 0; r < 8; ++r) pf.f[r] = acc[io][it][r] + bb;
        *(float4*)(orow + ibase)     = pf.q[0];
        *(float4*)(orow + ibase + 4) = pf.q[1];
      }
    } else {
      bf16_t* orow = hout + ((size_t)b * Sp + tp) * Nc + i0;
#pragma unroll
      for (int io = 0; io < 4; ++io) {
        const int ibase = io * 16 + 8 * half;
        Pack8 pk;
#pragma unroll
        for (int r = 0; r < 8; ++r) pk.e[r] = f2bf(acc[io][it][r] + bb);
        *(uint4*)(orow + ibase) = pk.q;
      }
    }
  }
}

// ---------------------------------------------------------------------------
extern "C" void kernel_launch(void* const* d_in, const int* in_sizes, int n_in,
                              void* d_out, int out_size, void* d_ws, size_t ws_size,
                              hipStream_t stream) {
  (void)in_sizes; (void)n_in; (void)out_size; (void)ws_size;
  const float* x      = (const float*)d_in[0];
  const float* c1w    = (const float*)d_in[1];
  const float* c1b    = (const float*)d_in[2];
  const float* c2w    = (const float*)d_in[3];
  const float* c2b    = (const float*)d_in[4];
  const float* gcw    = (const float*)d_in[5];
  const float* gcbias = (const float*)d_in[6];
  const float* logits = (const float*)d_in[7];
  const float* gnoise = (const float*)d_in[8];

  auto align256 = [](size_t v) { return (v + 255) & ~(size_t)255; };
  char* p = (char*)d_ws;
  bf16_t* h   = (bf16_t*)p; p += align256((size_t)Bz * Sp * Nc * sizeof(bf16_t));
  bf16_t* tmp = (bf16_t*)p; p += align256((size_t)Bz * Sp * Nc * sizeof(bf16_t));
  bf16_t* U   = (bf16_t*)p; p += align256((size_t)Bz * Nc * Sp * sizeof(bf16_t));
  bf16_t* w1k = (bf16_t*)p; p += align256((size_t)Lx * Kw * Nc * Nc * sizeof(bf16_t));
  bf16_t* w2k = (bf16_t*)p; p += align256((size_t)Lx * Kw * Nc * Nc * sizeof(bf16_t));
  bf16_t* gcT = (bf16_t*)p; p += align256((size_t)Sp * Sp * sizeof(bf16_t));
  bf16_t* Am  = (bf16_t*)p; p += align256((size_t)Lx * Nc * Nc * sizeof(bf16_t));

  const int BSN = Bz * Sp * Nc;
  cvt_f32_to_bf16<<<(BSN + 255) / 256, 256, 0, stream>>>(x, h, BSN);
  const int WTOT = Lx * Kw * Nc * Nc;
  prep_conv_w<<<(WTOT + 255) / 256, 256, 0, stream>>>(c1w, w1k, WTOT);
  prep_conv_w<<<(WTOT + 255) / 256, 256, 0, stream>>>(c2w, w2k, WTOT);
  const int GTOT = Sp * Sp;
  prep_gcT<<<(GTOT + 255) / 256, 256, 0, stream>>>(gcw, gcT, GTOT);
  const int ATOT = Lx * Nc * Nc;
  prep_adj<<<(ATOT + 255) / 256, 256, 0, stream>>>(logits, gnoise, Am, ATOT);

  dim3 blk(32, 1, 1);                 // one wave32 per block; EXEC all-ones for WMMA
  dim3 gconv(Sp / 32, Nc / 64, Bz);   // conv: M=o tiles of 64, N=t tiles of 32
  dim3 gagg (Sp / 64, Nc / 32, Bz);   // aggregate: M=t tiles of 64, N=i tiles of 32
  dim3 gmix (Sp / 32, Nc / 64, Bz);   // mix: M=i tiles of 64, N=t' tiles of 32

  for (int l = 0; l < Lx; ++l) {
    const int dil = 1 << l;
    const bf16_t* w1l = w1k + (size_t)l * Kw * Nc * Nc;
    const bf16_t* w2l = w2k + (size_t)l * Kw * Nc * Nc;
    // temporal block: relu(conv1) -> tmp ; relu(conv2 + residual h) -> h (in-place safe)
    conv_wmma<false><<<gconv, blk, 0, stream>>>(h, w1l, c1b + l * Nc, h, tmp, dil);
    conv_wmma<true ><<<gconv, blk, 0, stream>>>(tmp, w2l, c2b + l * Nc, h, h, dil);
    // graph conv: node mixing by hard-Gumbel A/N, then temporal mixing by gc_weight + bias
    gcn_aggregate<<<gagg, blk, 0, stream>>>(h, Am + (size_t)l * Nc * Nc, U);
    if (l == Lx - 1)
      gcn_mix<true ><<<gmix, blk, 0, stream>>>(U, gcT, gcbias, h, (float*)d_out);
    else
      gcn_mix<false><<<gmix, blk, 0, stream>>>(U, gcT, gcbias, h, (float*)d_out);
  }
}
